// StreamGQASelfAttentionRelPos_1632087573174
// MI455X (gfx1250) — matrix operand order repaired and checked by reading, not compiled, not run on hardware
//
#include <hip/hip_runtime.h>
#include <hip/hip_bf16.h>

// ---------- types ----------
typedef __attribute__((ext_vector_type(16))) __bf16 v16bf;
typedef __attribute__((ext_vector_type(8)))  float  v8f;

union V16BF { v16bf v; unsigned short u[16]; };

__device__ __forceinline__ unsigned short f2bf(float f) {
  union { __bf16 b; unsigned short u; } x;
  x.b = (__bf16)f;   // hardware cvt (RTNE) on gfx1250
  return x.u;
}

__device__ __forceinline__ v8f v8f_zero() {
  v8f z;
#pragma unroll
  for (int i = 0; i < 8; ++i) z[i] = 0.0f;
  return z;
}

__device__ __forceinline__ v8f wmma_bf16(v16bf a, v16bf b, v8f c) {
  return __builtin_amdgcn_wmma_f32_16x16x32_bf16(
      /*neg_a=*/false, a, /*neg_b=*/false, b,
      /*c_mod=*/(short)0, c, /*reuse_a=*/false, /*reuse_b=*/false);
}

// ---------- WMMA operand loaders (16x16x32 bf16 layouts, wave32) ----------
// A (16xK tile, bf16): lanes 0-15 row M=lane hold K {0..7, 16..23};
// lanes 16-31 row M=lane-16 hold K {8..15, 24..31}.
__device__ __forceinline__ v16bf load_a_f32(const float* A, int m0, int k0, int lda) {
  int lane = threadIdx.x & 31, half = lane >> 4, r = lane & 15;
  const float* row = A + (size_t)(m0 + r) * lda + k0;
  int s1 = half * 8, s2 = 16 + half * 8;
  V16BF o;
#pragma unroll
  for (int i = 0; i < 8; ++i) {
    o.u[i]     = f2bf(row[s1 + i]);
    o.u[8 + i] = f2bf(row[s2 + i]);
  }
  return o.v;
}

__device__ __forceinline__ v16bf load_a_bf16(const unsigned short* A, int m0, int k0, int lda) {
  int lane = threadIdx.x & 31, half = lane >> 4, r = lane & 15;
  const unsigned short* row = A + (size_t)(m0 + r) * lda + k0;
  int s1 = half * 8, s2 = 16 + half * 8;
  V16BF o;
#pragma unroll
  for (int i = 0; i < 8; ++i) {
    o.u[i]     = row[s1 + i];
    o.u[8 + i] = row[s2 + i];
  }
  return o.v;
}

// B (Kx16 tile): column N = lane&15; lanes 0-15 hold K {0..7, 16..23},
// lanes 16-31 hold K {8..15, 24..31}. Source rows are N-major with K contiguous.
__device__ __forceinline__ v16bf load_b_bf16(const unsigned short* Bm, int n0, int k0, int ld) {
  int lane = threadIdx.x & 31, half = lane >> 4, r = lane & 15;
  const unsigned short* p = Bm + (size_t)(n0 + r) * ld + k0 + half * 8;
  V16BF o;
#pragma unroll
  for (int i = 0; i < 8; ++i) {
    o.u[i]     = p[i];
    o.u[8 + i] = p[16 + i];
  }
  return o.v;
}

// ---------- elementwise f32 -> bf16 with zero padding ----------
__global__ __launch_bounds__(256) void to_bf16_pad(const float* __restrict__ src,
                                                   unsigned short* __restrict__ dst,
                                                   int n_src, int n_dst) {
  int i = blockIdx.x * 256 + threadIdx.x;
  if (i < n_dst) dst[i] = f2bf(i < n_src ? src[i] : 0.0f);
}

// ---------- unified projection GEMM: C[M,N] = alpha * A[M,K] @ W[N,K]^T ----------
// W is pre-converted bf16, row-major [N,K] (padded rows are zero).
// MODE: 0=Q(->qs_bf per-head), 1=K(->new_cache_k x4 + kh_bf),
//       2=V(->new_cache_v x4 + vt_bf transposed), 3 = plain f32 row-major (ld=N).
template <int MODE, bool ABF16>
__global__ __launch_bounds__(256) void wmma_gemm(
    const void* __restrict__ Av, const unsigned short* __restrict__ Wbf,
    int N, int K, int lda, float alpha,
    float* __restrict__ out_f, unsigned short* __restrict__ out_b)
{
  int wave = threadIdx.x >> 5;
  int lane = threadIdx.x & 31, half = lane >> 4, r = lane & 15;
  int m0 = blockIdx.x * 128 + wave * 16;
  int n0 = blockIdx.y * 64;

  v8f acc[4];
#pragma unroll
  for (int j = 0; j < 4; ++j) acc[j] = v8f_zero();

  for (int k0 = 0; k0 < K; k0 += 32) {
    v16bf a = ABF16 ? load_a_bf16((const unsigned short*)Av, m0, k0, lda)
                    : load_a_f32((const float*)Av, m0, k0, lda);
#pragma unroll
    for (int j = 0; j < 4; ++j) {
      v16bf b = load_b_bf16(Wbf, n0 + j * 16, k0, K);
      acc[j] = wmma_bf16(a, b, acc[j]);
    }
  }

  // epilogue; C layout: element (m0 + c + 8*half, n0 + j*16 + (lane&15))
#pragma unroll
  for (int j = 0; j < 4; ++j) {
#pragma unroll
    for (int c = 0; c < 8; ++c) {
      int m = m0 + c + 8 * half;
      int n = n0 + j * 16 + r;
      float val = acc[j][c] * alpha;
      if (MODE == 0) {
        int b = m >> 10, t = m & 1023, h = n >> 6, dd = n & 63;
        out_b[(((size_t)(b * 16 + h) * 1024 + t) << 6) + dd] = f2bf(val);
      } else if (MODE == 1) {
        int b = m >> 10, t = m & 1023, g = n >> 6, dd = n & 63;
#pragma unroll
        for (int rep = 0; rep < 4; ++rep)
          out_f[(((size_t)((b * 16 + g * 4 + rep) * 64 + dd)) << 10) + t] = val;
        out_b[(((size_t)(b * 4 + g) * 1024 + t) << 6) + dd] = f2bf(val);
      } else if (MODE == 2) {
        int b = m >> 10, t = m & 1023, g = n >> 6, dd = n & 63;
#pragma unroll
        for (int rep = 0; rep < 4; ++rep)
          out_f[(((size_t)((b * 16 + g * 4 + rep) * 1024 + t)) << 6) + dd] = val;
        // transposed bf16: vt_bf[bg, dd, t]
        out_b[(((size_t)(b * 4 + g) * 64 + dd) << 10) + t] = f2bf(val);
      } else {
        out_f[(size_t)m * N + n] = val;
      }
    }
  }
}

// ---------- cache K transpose+convert: [bh, 64, 1024] f32 -> [bh, 1024, 64] bf16 ----------
__global__ __launch_bounds__(256) void stage_kc(const float* __restrict__ ck,
                                                unsigned short* __restrict__ kc_bf) {
  __shared__ float tile[64][65];
  int bh = blockIdx.x;
  int s0 = blockIdx.y * 64;
  const float* src = ck + (size_t)bh * 64 * 1024;
  for (int e = threadIdx.x; e < 4096; e += 256) {
    int dd = e >> 6, sl = e & 63;
    tile[dd][sl] = src[(size_t)dd * 1024 + s0 + sl];
  }
  __syncthreads();
  unsigned short* dst = kc_bf + ((size_t)bh * 1024 + s0) * 64;
  for (int e = threadIdx.x; e < 4096; e += 256) {
    int sl = e >> 6, dd = e & 63;
    dst[(size_t)sl * 64 + dd] = f2bf(tile[dd][sl]);
  }
}

// ---------- cache V transpose+convert: [bh, 1024, 64] f32 -> [bh, 64, 1024] bf16 ----------
__global__ __launch_bounds__(256) void stage_vc(const float* __restrict__ cv,
                                                unsigned short* __restrict__ vtc_bf) {
  __shared__ float tile[64][65];
  int bh = blockIdx.x;
  int s0 = blockIdx.y * 64;
  const float* src = cv + ((size_t)bh * 1024 + s0) * 64;
  for (int e = threadIdx.x; e < 4096; e += 256) {
    int sl = e >> 6, dd = e & 63;
    tile[sl][dd] = src[(size_t)sl * 64 + dd];
  }
  __syncthreads();
  unsigned short* dst = vtc_bf + (size_t)bh * 64 * 1024;
  for (int e = threadIdx.x; e < 4096; e += 256) {
    int dd = e >> 6, sl = e & 63;
    dst[(size_t)dd * 1024 + s0 + sl] = f2bf(tile[sl][dd]);
  }
}

// ---------- flash attention: each wave owns 16 query rows, streams S in 64-chunks ----------
__global__ __launch_bounds__(256) void attn_kernel(
    const unsigned short* __restrict__ qs_bf, const unsigned short* __restrict__ kc_bf,
    const unsigned short* __restrict__ kh_bf, const unsigned short* __restrict__ vtc_bf,
    const unsigned short* __restrict__ vt_bf, const float* __restrict__ proj,
    unsigned short* __restrict__ o_bf)
{
  __shared__ unsigned short Vt[64 * 64];        // [dd][s_local] bf16
  __shared__ unsigned short Plds[8 * 16 * 64];  // per-wave P (16 x 64) bf16

  int bh = blockIdx.x, b = bh >> 4, h = bh & 15, bg = b * 4 + (h >> 2);
  int wave = threadIdx.x >> 5, lane = threadIdx.x & 31, half = lane >> 4, r = lane & 15;
  int t0 = blockIdx.y * 128 + wave * 16;

  const unsigned short* qrows = qs_bf + (size_t)bh * 1024 * 64;
  v16bf a0 = load_a_bf16(qrows, t0, 0, 64);
  v16bf a1 = load_a_bf16(qrows, t0, 32, 64);

  v8f oacc[4];
#pragma unroll
  for (int j = 0; j < 4; ++j) oacc[j] = v8f_zero();
  v8f mrow, lrow = v8f_zero();
#pragma unroll
  for (int c = 0; c < 8; ++c) mrow[c] = -1e30f;

  unsigned short* pb = &Plds[wave * 16 * 64];
  const float* projrow = proj + (size_t)bh * 1024 * 256;

  for (int s0 = 0; s0 < 2048; s0 += 64) {
    // V chunk source: pre-transposed bf16 [*, dd, s], row stride 1024
    const unsigned short* vsrc = (s0 < 1024)
        ? vtc_bf + (size_t)bh * 64 * 1024 + s0
        : vt_bf + (size_t)bg * 64 * 1024 + (s0 - 1024);
    const unsigned short* ksrc = (s0 < 1024)
        ? kc_bf + ((size_t)bh * 1024 + s0) * 64
        : kh_bf + ((size_t)bg * 1024 + (s0 - 1024)) * 64;

    __syncthreads();  // previous iteration's Vt reads complete
    // async copy 64 rows x 128B into LDS (ASYNCcnt-tracked, b128 per lane)
#pragma unroll
    for (int it = 0; it < 2; ++it) {
      int e = it * 256 + threadIdx.x;     // 512 x 16-byte chunks
      int dd = e >> 3, s8 = (e & 7) << 3;
      unsigned long long g =
          (unsigned long long)(uintptr_t)(vsrc + (size_t)dd * 1024 + s8);
      unsigned lds = (unsigned)(uintptr_t)(&Vt[dd * 64 + s8]);
      asm volatile("global_load_async_to_lds_b128 %0, %1, off"
                   :: "v"(lds), "v"(g) : "memory");
    }
    // prefetch next K chunk while async copy is in flight
    if (s0 + 64 < 2048) {
      const unsigned short* knext = (s0 + 64 < 1024)
          ? kc_bf + ((size_t)bh * 1024 + s0 + 64) * 64
          : kh_bf + ((size_t)bg * 1024 + (s0 + 64 - 1024)) * 64;
      __builtin_prefetch(knext + (size_t)threadIdx.x * 16, 0, 3);
    }
    asm volatile("s_wait_asynccnt 0x0" ::: "memory");
    __syncthreads();

    // scores: S = qs(16x64) @ K_chunk(64x64)^T
    v8f sc[4];
#pragma unroll
    for (int j = 0; j < 4; ++j) {
      v16bf b0 = load_b_bf16(ksrc, j * 16, 0, 64);
      v16bf b1 = load_b_bf16(ksrc, j * 16, 32, 64);
      v8f acc = v8f_zero();
      acc = wmma_bf16(a0, b0, acc);
      acc = wmma_bf16(a1, b1, acc);
      sc[j] = acc;
    }

    // relative-position bias gather
#pragma unroll
    for (int j = 0; j < 4; ++j) {
#pragma unroll
      for (int c = 0; c < 8; ++c) {
        int tq = t0 + c + 8 * half;
        int s = s0 + j * 16 + r;
        int raw = tq - s + 1024;
        int idx = (raw < -127 ? -127 : (raw > 127 ? 127 : raw)) + 127;
        sc[j][c] += projrow[(size_t)tq * 256 + idx];
      }
    }

    // online softmax (rows live in half-wave: row = c + 8*half, cols across 16 lanes)
    v8f mx = sc[0];
#pragma unroll
    for (int j = 1; j < 4; ++j)
#pragma unroll
      for (int c = 0; c < 8; ++c) mx[c] = fmaxf(mx[c], sc[j][c]);
#pragma unroll
    for (int msk = 1; msk <= 8; msk <<= 1)
#pragma unroll
      for (int c = 0; c < 8; ++c) mx[c] = fmaxf(mx[c], __shfl_xor(mx[c], msk));

    v8f mnew, al;
#pragma unroll
    for (int c = 0; c < 8; ++c) {
      mnew[c] = fmaxf(mrow[c], mx[c]);
      al[c] = __expf(mrow[c] - mnew[c]);
    }
    v8f rs = v8f_zero();
#pragma unroll
    for (int j = 0; j < 4; ++j)
#pragma unroll
      for (int c = 0; c < 8; ++c) {
        float p = __expf(sc[j][c] - mnew[c]);
        sc[j][c] = p;
        rs[c] += p;
      }
#pragma unroll
    for (int msk = 1; msk <= 8; msk <<= 1)
#pragma unroll
      for (int c = 0; c < 8; ++c) rs[c] += __shfl_xor(rs[c], msk);
#pragma unroll
    for (int c = 0; c < 8; ++c) {
      lrow[c] = lrow[c] * al[c] + rs[c];
      mrow[c] = mnew[c];
    }
#pragma unroll
    for (int j = 0; j < 4; ++j)
#pragma unroll
      for (int c = 0; c < 8; ++c) oacc[j][c] *= al[c];

    // stage P (C layout -> row-major bf16 in per-wave LDS), then O += P @ V_chunk
#pragma unroll
    for (int j = 0; j < 4; ++j)
#pragma unroll
      for (int c = 0; c < 8; ++c)
        pb[(c + 8 * half) * 64 + j * 16 + r] = f2bf(sc[j][c]);

    v16bf p0 = load_a_bf16(pb, 0, 0, 64);
    v16bf p1 = load_a_bf16(pb, 0, 32, 64);
#pragma unroll
    for (int j = 0; j < 4; ++j) {
      v16bf bv0 = load_b_bf16(Vt, j * 16, 0, 64);
      v16bf bv1 = load_b_bf16(Vt, j * 16, 32, 64);
      oacc[j] = wmma_bf16(p0, bv0, oacc[j]);
      oacc[j] = wmma_bf16(p1, bv1, oacc[j]);
    }
  }

  // normalize + store o (bf16, [b, t, h*64+dd]) for the Wo GEMM
#pragma unroll
  for (int j = 0; j < 4; ++j) {
#pragma unroll
    for (int c = 0; c < 8; ++c) {
      int tq = t0 + c + 8 * half;
      int n = h * 64 + j * 16 + r;
      float val = oacc[j][c] / lrow[c];
      o_bf[((size_t)b * 1024 + tq) * 1024 + n] = f2bf(val);
    }
  }
}

// ---------- launcher ----------
extern "C" void kernel_launch(void* const* d_in, const int* in_sizes, int n_in,
                              void* d_out, int out_size, void* d_ws, size_t ws_size,
                              hipStream_t stream) {
  (void)in_sizes; (void)n_in; (void)out_size; (void)ws_size;
  const float* q   = (const float*)d_in[0];
  const float* k   = (const float*)d_in[1];
  const float* v   = (const float*)d_in[2];
  const float* ck  = (const float*)d_in[3];
  const float* cv  = (const float*)d_in[4];
  const float* Wq  = (const float*)d_in[5];
  const float* Wk  = (const float*)d_in[6];
  const float* Wv  = (const float*)d_in[7];
  const float* Wo  = (const float*)d_in[8];
  const float* rel = (const float*)d_in[9];

  float* out = (float*)d_out;
  float* nck = out + (size_t)2 * 1024 * 1024;  // new_cache_k [32,64,1024]
  float* ncv = nck + (size_t)2 * 1024 * 1024;  // new_cache_v [32,1024,64]

#define WS_MB(x) ((size_t)(x) << 20)
  char* ws = (char*)d_ws;
  unsigned short* qs_bf  = (unsigned short*)(ws);                          // 4 MB [32,1024,64]
  unsigned short* kh_bf  = (unsigned short*)(ws + WS_MB(4));               // 1 MB [8,1024,64]
  unsigned short* vt_bf  = (unsigned short*)(ws + WS_MB(5));               // 1 MB [8,64,1024]
  unsigned short* kc_bf  = (unsigned short*)(ws + WS_MB(6));               // 4 MB [32,1024,64]
  unsigned short* o_bf   = (unsigned short*)(ws + WS_MB(10));              // 4 MB [2048,1024]
  unsigned short* vtc_bf = (unsigned short*)(ws + WS_MB(14));              // 4 MB [32,64,1024]
  unsigned short* wq_bf  = (unsigned short*)(ws + WS_MB(18));              // 2 MB
  unsigned short* wo_bf  = (unsigned short*)(ws + WS_MB(20));              // 2 MB
  unsigned short* wk_bf  = (unsigned short*)(ws + WS_MB(22));              // 0.5 MB
  unsigned short* wv_bf  = (unsigned short*)(ws + WS_MB(22) + (512 << 10));// 0.5 MB
  unsigned short* rel_bf = (unsigned short*)(ws + WS_MB(23));              // 32 KB [256,64]
  float*          proj   = (float*)(ws + WS_MB(24));                       // 32 MB [32,1024,256]

  dim3 blk(256);
  // weight conversions (bf16, rel padded to 256 rows)
  to_bf16_pad<<<dim3(4096), blk, 0, stream>>>(Wq, wq_bf, 1048576, 1048576);
  to_bf16_pad<<<dim3(1024), blk, 0, stream>>>(Wk, wk_bf, 262144, 262144);
  to_bf16_pad<<<dim3(1024), blk, 0, stream>>>(Wv, wv_bf, 262144, 262144);
  to_bf16_pad<<<dim3(4096), blk, 0, stream>>>(Wo, wo_bf, 1048576, 1048576);
  to_bf16_pad<<<dim3(64),   blk, 0, stream>>>(rel, rel_bf, 16320, 16384);

  // qs = (q @ Wq^T) * 1/sqrt(64), per-head bf16
  wmma_gemm<0, false><<<dim3(16, 16), blk, 0, stream>>>(q, wq_bf, 1024, 1024, 1024,
                                                        0.125f, nullptr, qs_bf);
  // kh -> new_cache_k (x4 per group) + bf16 copy [bg, t, dd]
  wmma_gemm<1, false><<<dim3(16, 4), blk, 0, stream>>>(k, wk_bf, 256, 1024, 1024,
                                                       1.0f, nck, kh_bf);
  // vh -> new_cache_v (x4 per group) + transposed bf16 copy [bg, dd, t]
  wmma_gemm<2, false><<<dim3(16, 4), blk, 0, stream>>>(v, wv_bf, 256, 1024, 1024,
                                                       1.0f, ncv, vt_bf);
  // rel projection: proj[bh,t,i] = qs . rel_emb[i]  (N padded to 256, row 255 zero)
  wmma_gemm<3, true><<<dim3(256, 4), blk, 0, stream>>>(qs_bf, rel_bf, 256, 64, 64,
                                                       1.0f, proj, nullptr);
  // caches -> bf16 staging layouts
  stage_kc<<<dim3(32, 16), blk, 0, stream>>>(ck, kc_bf);
  stage_vc<<<dim3(32, 16), blk, 0, stream>>>(cv, vtc_bf);
  // flash attention over S=2048
  attn_kernel<<<dim3(32, 8), blk, 0, stream>>>(qs_bf, kc_bf, kh_bf, vtc_bf, vt_bf,
                                               proj, o_bf);
  // out = o @ Wo^T
  wmma_gemm<3, true><<<dim3(16, 16), blk, 0, stream>>>(o_bf, wo_bf, 1024, 1024, 1024,
                                                       1.0f, out, nullptr);
#undef WS_MB
}
